// BasicTransBlock_20478404067807
// MI455X (gfx1250) — matrix operand order repaired and checked by reading, not compiled, not run on hardware
//
#include <hip/hip_runtime.h>
#include <hip/hip_bf16.h>

// ---------------------------------------------------------------------------
// Types for CDNA5 WMMA (wave32, v_wmma_f32_16x16x32_bf16)
// ---------------------------------------------------------------------------
typedef __attribute__((ext_vector_type(16))) __bf16 v16bf;
typedef __attribute__((ext_vector_type(8)))  __bf16 v8bf;
typedef __attribute__((ext_vector_type(8)))  float  v8f;

union Frag16 { v16bf v; v8bf h[2]; };

#define NB 8
#define NC 256
#define NH 64
#define NW 64
#define NHW 4096
#define NHEADS 8
#define NDH 32
#define NRS 16
#define NRS2 256
#define NC3 768
#define BN_EPS 1e-5f

__device__ __forceinline__ unsigned short f2bf(float f) {
    unsigned u = __builtin_bit_cast(unsigned int, f);
    unsigned r = u + 0x7fffu + ((u >> 16) & 1u);
    return (unsigned short)(r >> 16);
}

// ---------------------------------------------------------------------------
// Async staging: copy a contiguous panel global -> LDS via the ASYNCcnt path
// (global_load_async_to_lds_b128; ISA 15.18.3 op 98). LDS byte address is the
// low 32 bits of the generic pointer.
// ---------------------------------------------------------------------------
__device__ __forceinline__ void stage_panel_async(unsigned short* Bsh,
                                                  const unsigned short* gsrc,
                                                  int bytes, int tid, int nthreads) {
    for (int off = tid * 16; off < bytes; off += nthreads * 16) {
        unsigned lds = (unsigned)(size_t)((char*)Bsh + off);
        const char* g = (const char*)gsrc + off;
        asm volatile("global_load_async_to_lds_b128 %0, %1, off"
                     :: "v"(lds), "v"(g) : "memory");
    }
    asm volatile("s_wait_asynccnt 0x0" ::: "memory");
}

// ---------------------------------------------------------------------------
// One 16M x 64N row-tile: A from global (row-major [16][NC]), B panel from LDS
// ([64 n-rows][NC k]); A fragment loaded once per K-step, 4 WMMAs issued.
// Per-lane fragment addressing follows ISA 7.12.2 (wave32).
// ---------------------------------------------------------------------------
__device__ __forceinline__ void wmma_row_4n(const unsigned short* __restrict__ Arow,
                                            const unsigned short* __restrict__ Bsh,
                                            unsigned lane, v8f acc[4]) {
    const unsigned short* aptr = Arow + (lane & 15u) * NC + (lane >> 4) * 8;
    const unsigned short* b0   = Bsh  + (lane & 15u) * NC + (lane >> 4) * 8;
    __builtin_prefetch((const void*)(aptr + 16 * NC), 0, 0);   // next M tile
    for (int kk = 0; kk < NC; kk += 32) {
        Frag16 fa;
        fa.h[0] = *(const v8bf*)(aptr + kk);
        fa.h[1] = *(const v8bf*)(aptr + kk + 16);
#pragma unroll
        for (int nt = 0; nt < 4; nt++) {
            Frag16 fb;
            const unsigned short* bp = b0 + nt * 16 * NC;
            fb.h[0] = *(const v8bf*)(bp + kk);
            fb.h[1] = *(const v8bf*)(bp + kk + 16);
            acc[nt] = __builtin_amdgcn_wmma_f32_16x16x32_bf16(false, fa.v, false, fb.v,
                                                              (short)0, acc[nt], false, false);
        }
    }
}

// ---------------------------------------------------------------------------
// K0: fp32 -> bf16 weight conversion for the three pointwise GEMMs
// ---------------------------------------------------------------------------
__global__ void k0_convert_weights(const float* __restrict__ qkv_pw,
                                   const float* __restrict__ out_pw,
                                   const float* __restrict__ mlp_w,
                                   unsigned short* __restrict__ wqkv,
                                   unsigned short* __restrict__ wout,
                                   unsigned short* __restrict__ wmlp) {
    int i = blockIdx.x * 256 + threadIdx.x;
    if (i < NC3 * NC) wqkv[i] = f2bf(qkv_pw[i]);
    if (i < NC * NC) { wout[i] = f2bf(out_pw[i]); wmlp[i] = f2bf(mlp_w[i]); }
}

// ---------------------------------------------------------------------------
// K1/K5: optional BN then depthwise 3x3; output bf16 in [b][pixel][c] layout
// ---------------------------------------------------------------------------
__global__ void k1_bn_dwconv(const float* __restrict__ in,    // [B][C][64][64]
                             const float* __restrict__ gamma, const float* __restrict__ beta,
                             const float* __restrict__ mean,  const float* __restrict__ var,
                             int use_bn,
                             const float* __restrict__ dw,    // [C][1][3][3]
                             unsigned short* __restrict__ out)// [B][4096][C] bf16
{
    int idx = blockIdx.x * blockDim.x + threadIdx.x;     // over B*C*HW
    int xw = idx & 63, y = (idx >> 6) & 63, c = (idx >> 12) & 255, b = idx >> 20;
    float sc = 1.f, sh = 0.f;
    if (use_bn) {
        float inv = gamma[c] * rsqrtf(var[c] + BN_EPS);
        sc = inv; sh = beta[c] - mean[c] * inv;
    }
    const float* p = in + (((size_t)b * NC + c) << 12);
    const float* w = dw + c * 9;
    float acc = 0.f;
#pragma unroll
    for (int ky = -1; ky <= 1; ky++) {
        int yy = y + ky;
        if (yy < 0 || yy > 63) continue;
#pragma unroll
        for (int kx = -1; kx <= 1; kx++) {
            int xx = xw + kx;
            if (xx < 0 || xx > 63) continue;
            acc += (p[(yy << 6) + xx] * sc + sh) * w[(ky + 1) * 3 + (kx + 1)];
        }
    }
    out[((size_t)b * NHW + (y << 6) + xw) * NC + c] = f2bf(acc);
}

// ---------------------------------------------------------------------------
// K2: qkv pointwise GEMM  M=B*4096, N=768, K=256.
// Block: 4 waves x (16M x 64N); B panel (64x256 bf16 = 32KB) async-staged in LDS.
// D-scatter splits q/k/v.
// ---------------------------------------------------------------------------
__global__ __launch_bounds__(128) void k2_qkv_gemm(
        const unsigned short* __restrict__ A,   // [B*4096][256] bf16
        const unsigned short* __restrict__ Wq,  // [768][256]    bf16
        unsigned short* __restrict__ q,         // [B*8][4096][32] bf16
        float* __restrict__ kf,                 // [B][256][4096]
        float* __restrict__ vf)                 // [B][256][4096]
{
    __shared__ unsigned short Bsh[64 * NC];     // 32 KB
    unsigned lane = threadIdx.x & 31, wave = threadIdx.x >> 5;
    stage_panel_async(Bsh, Wq + (size_t)blockIdx.y * 64 * NC, 64 * NC * 2,
                      threadIdx.x, 128);
    __syncthreads();

    int rowbase = (blockIdx.x * 4 + wave) * 16;             // pixel-row tile
    v8f acc[4] = {{}, {}, {}, {}};
    wmma_row_4n(A + (size_t)rowbase * NC, Bsh, lane, acc);

    int n = lane & 15, m0 = (lane >> 4) * 8;
#pragma unroll
    for (int nt = 0; nt < 4; nt++) {
        int oc = blockIdx.y * 64 + nt * 16 + n;
#pragma unroll
        for (int r = 0; r < 8; r++) {
            int row = rowbase + m0 + r;
            int b = row >> 12, pix = row & 4095;
            float val = acc[nt][r];
            if (oc < NC) {
                int head = oc & 7, d = oc >> 3;
                q[(((size_t)(b * NHEADS + head) * NHW + pix) << 5) + d] = f2bf(val);
            } else if (oc < 2 * NC) {
                kf[((size_t)b * NC + (oc - NC)) * NHW + pix] = val;
            } else {
                vf[((size_t)b * NC + (oc - 2 * NC)) * NHW + pix] = val;
            }
        }
    }
}

// ---------------------------------------------------------------------------
// K3: align-corners bilinear 64x64 -> 16x16 for k and v, head-split bf16 out
//     k -> [bh][j][d]  (WMMA-B layout for q@k^T),  v -> [bh][d][j] (for P@v)
// ---------------------------------------------------------------------------
__global__ void k3_interp(const float* __restrict__ kf, const float* __restrict__ vf,
                          unsigned short* __restrict__ ks, unsigned short* __restrict__ vs) {
    int idx = blockIdx.x * 256 + threadIdx.x;        // over 2*B*C*256
    int which = idx >= NB * NC * NRS2;
    int i = which ? idx - NB * NC * NRS2 : idx;
    int ox = i & 15, oy = (i >> 4) & 15, c = (i >> 8) & 255, b = i >> 16;
    const float s = 63.f / 15.f;
    float ysf = oy * s, xsf = ox * s;
    int y0 = (int)floorf(ysf), x0 = (int)floorf(xsf);
    int y1 = min(y0 + 1, 63), x1 = min(x0 + 1, 63);
    float wy = ysf - y0, wx = xsf - x0;
    const float* p = (which ? vf : kf) + (((size_t)b * NC + c) << 12);
    float v00 = p[(y0 << 6) + x0], v01 = p[(y0 << 6) + x1];
    float v10 = p[(y1 << 6) + x0], v11 = p[(y1 << 6) + x1];
    float val = (v00 * (1 - wy) + v10 * wy) * (1 - wx) + (v01 * (1 - wy) + v11 * wy) * wx;
    int head = c & 7, d = c >> 3, j = oy * 16 + ox;
    size_t bh = (size_t)b * NHEADS + head;
    if (!which) ks[((bh * NRS2 + j) << 5) + d] = f2bf(val);
    else        vs[((bh << 5) + d) * NRS2 + j] = f2bf(val);
}

// ---------------------------------------------------------------------------
// K4: fused attention. Per wave: 16 query rows for one (b,head):
//     S = q@k^T (16 WMMA) + rel-bias, *scale -> LDS; wave-parallel row softmax;
//     P(bf16, LDS) @ v (16 WMMA) -> o [B][C][HW] f32.
// ---------------------------------------------------------------------------
__global__ __launch_bounds__(64) void k4_attention(
        const unsigned short* __restrict__ q,    // [B*8][4096][32]
        const unsigned short* __restrict__ ks,   // [B*8][256][32]
        const unsigned short* __restrict__ vs,   // [B*8][32][256]
        const float* __restrict__ rel_table,     // [961][8]
        float* __restrict__ o)                   // [B][256][4096]
{
    __shared__ float          S[2][16][NRS2];
    __shared__ unsigned short P[2][16][NRS2];
    unsigned lane = threadIdx.x & 31, wave = threadIdx.x >> 5;
    int bh = blockIdx.y, b = bh >> 3, head = bh & 7;
    int rowbase = (blockIdx.x * 2 + wave) * 16;             // query pixel base
    const unsigned short* qb = q  + ((size_t)bh * NHW + rowbase) * NDH;
    const unsigned short* kb = ks + (size_t)bh * NRS2 * NDH;
    const unsigned short* vb = vs + (size_t)bh * NDH * NRS2;

    int n = lane & 15, m0 = (lane >> 4) * 8;
    // q A-fragment (16x32, K = DH): invariant across the 16 key tiles
    Frag16 fa;
    const unsigned short* aq = qb + (lane & 15) * NDH + (lane >> 4) * 8;
    fa.h[0] = *(const v8bf*)(aq);
    fa.h[1] = *(const v8bf*)(aq + 16);
    const float scale = 0.17677669529663687f;               // 32^-0.5

    for (int nt = 0; nt < 16; nt++) {
        Frag16 fb;
        const unsigned short* bp = kb + (size_t)(nt * 16 + (lane & 15)) * NDH + (lane >> 4) * 8;
        fb.h[0] = *(const v8bf*)(bp);
        fb.h[1] = *(const v8bf*)(bp + 16);
        v8f accS = {};
        accS = __builtin_amdgcn_wmma_f32_16x16x32_bf16(false, fa.v, false, fb.v,
                                                       (short)0, accS, false, false);
        int j = nt * 16 + n, jy = j >> 4, jx = j & 15;
#pragma unroll
        for (int r = 0; r < 8; r++) {
            int qrow = rowbase + m0 + r;
            int iy = (qrow >> 6) >> 2, ix = (qrow & 63) >> 2;
            int rel = (iy - jy + 15) * 31 + (ix - jx + 15);
            float bias = rel_table[rel * NHEADS + head];
            S[wave][m0 + r][j] = (accS[r] + bias) * scale;
        }
    }
    __syncthreads();

    // wave-parallel softmax: lane -> (row = lane&15, half = lane>>4 of 128 cols),
    // halves merged with one shfl_xor(16).
    {
        int rrow = lane & 15, hf = lane >> 4;
        float* srow = &S[wave][rrow][0];
        int j0 = hf * 128, j1 = j0 + 128;
        float mx = -1e30f;
        for (int j = j0; j < j1; j++) mx = fmaxf(mx, srow[j]);
        mx = fmaxf(mx, __shfl_xor(mx, 16, 32));
        float sum = 0.f;
        for (int j = j0; j < j1; j++) { float e = __expf(srow[j] - mx); srow[j] = e; sum += e; }
        sum += __shfl_xor(sum, 16, 32);
        float inv = 1.f / sum;
        for (int j = j0; j < j1; j++) P[wave][rrow][j] = f2bf(srow[j] * inv);
    }
    __syncthreads();

    const unsigned short* Pbase = &P[wave][0][0];
    for (int nt = 0; nt < 2; nt++) {
        v8f acc = {};
        const unsigned short* ap = Pbase + (lane & 15) * NRS2 + (lane >> 4) * 8;
        const unsigned short* bp = vb + (size_t)(nt * 16 + (lane & 15)) * NRS2 + (lane >> 4) * 8;
        for (int kk = 0; kk < NRS2; kk += 32) {
            Frag16 pa, pb;
            pa.h[0] = *(const v8bf*)(ap + kk);
            pa.h[1] = *(const v8bf*)(ap + kk + 16);
            pb.h[0] = *(const v8bf*)(bp + kk);
            pb.h[1] = *(const v8bf*)(bp + kk + 16);
            acc = __builtin_amdgcn_wmma_f32_16x16x32_bf16(false, pa.v, false, pb.v,
                                                          (short)0, acc, false, false);
        }
        int d = nt * 16 + n, c = d * NHEADS + head;
#pragma unroll
        for (int r = 0; r < 8; r++) {
            int qrow = rowbase + m0 + r;
            o[((size_t)b * NC + c) * NHW + qrow] = acc[r];
        }
    }
}

// ---------------------------------------------------------------------------
// K6: out-projection GEMM (N=K=256) with fused residual (+x), BN2+ReLU.
// Same 4-wave / 16Mx64N / async-B-panel structure as K2.
// ---------------------------------------------------------------------------
__global__ __launch_bounds__(128) void k6_out_gemm(
        const unsigned short* __restrict__ A,   // dwconv(o) [B*4096][256] bf16
        const unsigned short* __restrict__ W,   // out_pw    [256][256]    bf16
        const float* __restrict__ x,            // [B][256][4096]
        const float* __restrict__ g2, const float* __restrict__ b2,
        const float* __restrict__ m2, const float* __restrict__ v2,
        float* __restrict__ res,                // [B][256][4096]
        unsigned short* __restrict__ tb)        // relu(bn2) [B*4096][256] bf16
{
    __shared__ unsigned short Bsh[64 * NC];
    unsigned lane = threadIdx.x & 31, wave = threadIdx.x >> 5;
    stage_panel_async(Bsh, W + (size_t)blockIdx.y * 64 * NC, 64 * NC * 2,
                      threadIdx.x, 128);
    __syncthreads();

    int rowbase = (blockIdx.x * 4 + wave) * 16;
    v8f acc[4] = {{}, {}, {}, {}};
    wmma_row_4n(A + (size_t)rowbase * NC, Bsh, lane, acc);

    int n = lane & 15, m0 = (lane >> 4) * 8;
#pragma unroll
    for (int nt = 0; nt < 4; nt++) {
        int oc = blockIdx.y * 64 + nt * 16 + n;
        float inv = g2[oc] * rsqrtf(v2[oc] + BN_EPS);
        float sh  = b2[oc] - m2[oc] * inv;
#pragma unroll
        for (int r = 0; r < 8; r++) {
            int row = rowbase + m0 + r;
            int b = row >> 12, pix = row & 4095;
            size_t gi = ((size_t)b * NC + oc) * NHW + pix;
            float outv = acc[nt][r] + x[gi];
            res[gi] = outv;
            tb[(size_t)row * NC + oc] = f2bf(fmaxf(outv * inv + sh, 0.f));
        }
    }
}

// ---------------------------------------------------------------------------
// K7: MLP GEMM (N=K=256) with fused residual -> d_out [B][C][H][W] f32.
// ---------------------------------------------------------------------------
__global__ __launch_bounds__(128) void k7_mlp_gemm(
        const unsigned short* __restrict__ A,   // [B*4096][256] bf16
        const unsigned short* __restrict__ W,   // [256][256]    bf16
        const float* __restrict__ res,          // [B][256][4096]
        float* __restrict__ out)                // [B][256][4096]
{
    __shared__ unsigned short Bsh[64 * NC];
    unsigned lane = threadIdx.x & 31, wave = threadIdx.x >> 5;
    stage_panel_async(Bsh, W + (size_t)blockIdx.y * 64 * NC, 64 * NC * 2,
                      threadIdx.x, 128);
    __syncthreads();

    int rowbase = (blockIdx.x * 4 + wave) * 16;
    v8f acc[4] = {{}, {}, {}, {}};
    wmma_row_4n(A + (size_t)rowbase * NC, Bsh, lane, acc);

    int n = lane & 15, m0 = (lane >> 4) * 8;
#pragma unroll
    for (int nt = 0; nt < 4; nt++) {
        int oc = blockIdx.y * 64 + nt * 16 + n;
#pragma unroll
        for (int r = 0; r < 8; r++) {
            int row = rowbase + m0 + r;
            int b = row >> 12, pix = row & 4095;
            size_t gi = ((size_t)b * NC + oc) * NHW + pix;
            out[gi] = acc[nt][r] + res[gi];
        }
    }
}

// ---------------------------------------------------------------------------
extern "C" void kernel_launch(void* const* d_in, const int* in_sizes, int n_in,
                              void* d_out, int out_size, void* d_ws, size_t ws_size,
                              hipStream_t stream) {
    const float* x       = (const float*)d_in[0];
    const float* bn1_g   = (const float*)d_in[1];
    const float* bn1_b   = (const float*)d_in[2];
    const float* bn1_m   = (const float*)d_in[3];
    const float* bn1_v   = (const float*)d_in[4];
    const float* qkv_dw  = (const float*)d_in[5];
    const float* qkv_pw  = (const float*)d_in[6];
    const float* out_dw  = (const float*)d_in[7];
    const float* out_pw  = (const float*)d_in[8];
    const float* rel_tab = (const float*)d_in[9];
    const float* bn2_g   = (const float*)d_in[10];
    const float* bn2_b   = (const float*)d_in[11];
    const float* bn2_m   = (const float*)d_in[12];
    const float* bn2_v   = (const float*)d_in[13];
    const float* mlp_w   = (const float*)d_in[14];
    float* out = (float*)d_out;

    char* ws = (char*)d_ws;
    size_t off = 0;
    auto take = [&](size_t bytes) { char* p = ws + off; off += (bytes + 255) & ~(size_t)255; return p; };
    unsigned short* tmp1 = (unsigned short*)take((size_t)NB * NHW * NC * 2);   // also reused as tb
    unsigned short* wqkv = (unsigned short*)take((size_t)NC3 * NC * 2);
    unsigned short* wout = (unsigned short*)take((size_t)NC * NC * 2);
    unsigned short* wmlp = (unsigned short*)take((size_t)NC * NC * 2);
    unsigned short* qb   = (unsigned short*)take((size_t)NB * NHEADS * NHW * NDH * 2);
    float*          kf   = (float*)take((size_t)NB * NC * NHW * 4);            // reused as o
    float*          vf   = (float*)take((size_t)NB * NC * NHW * 4);            // reused as t2 (bf16)
    unsigned short* kss  = (unsigned short*)take((size_t)NB * NHEADS * NRS2 * NDH * 2);
    unsigned short* vss  = (unsigned short*)take((size_t)NB * NHEADS * NDH * NRS2 * 2);
    float*          res  = (float*)take((size_t)NB * NC * NHW * 4);
    float*          obuf = kf;                      // kf dead after K3
    unsigned short* t2   = (unsigned short*)vf;     // vf dead after K3
    unsigned short* tb   = tmp1;                    // tmp1 dead after K2

    // K0: weight conversion
    k0_convert_weights<<<NC3, 256, 0, stream>>>(qkv_pw, out_pw, mlp_w, wqkv, wout, wmlp);
    // K1: BN1 + depthwise qkv conv
    k1_bn_dwconv<<<(NB * NC * NHW) / 256, 256, 0, stream>>>(
        x, bn1_g, bn1_b, bn1_m, bn1_v, 1, qkv_dw, tmp1);
    // K2: qkv GEMM + scatter (block = 64M x 64N)
    k2_qkv_gemm<<<dim3((NB * NHW) / 64, NC3 / 64), 128, 0, stream>>>(tmp1, wqkv, qb, kf, vf);
    // K3: bilinear downsample k,v
    k3_interp<<<(2 * NB * NC * NRS2) / 256, 256, 0, stream>>>(kf, vf, kss, vss);
    // K4: fused attention
    k4_attention<<<dim3(NHW / 32, NB * NHEADS), 64, 0, stream>>>(qb, kss, vss, rel_tab, obuf);
    // K5: depthwise out conv (no BN)
    k1_bn_dwconv<<<(NB * NC * NHW) / 256, 256, 0, stream>>>(
        obuf, bn1_g, bn1_b, bn1_m, bn1_v, 0, out_dw, t2);
    // K6: out-projection GEMM + residual + BN2/ReLU
    k6_out_gemm<<<dim3((NB * NHW) / 64, NC / 64), 128, 0, stream>>>(
        t2, wout, x, bn2_g, bn2_b, bn2_m, bn2_v, res, tb);
    // K7: MLP GEMM + residual -> output
    k7_mlp_gemm<<<dim3((NB * NHW) / 64, NC / 64), 128, 0, stream>>>(tb, wmlp, res, out);
}
